// JointNetwork_37503654429039
// MI455X (gfx1250) — compile-verified, matrix operand order checked by reading, not run on hardware
//
#include <hip/hip_runtime.h>
#include <math.h>

typedef _Float16 half16 __attribute__((ext_vector_type(16)));
typedef _Float16 half8  __attribute__((ext_vector_type(8)));
typedef float    f32x8  __attribute__((ext_vector_type(8)));

#define JDIM 512
#define VDIM 1024
#define UDIM 64
#define TDIM 256
#define BDIM 4
#define BT   (BDIM * TDIM)
#define HS_STRIDE (JDIM + 8)   // 520 halves = 1040 B: 16B-aligned rows, 4-bank row skew

// ---------------- kernel 1: W_out fp32 -> fp16 ----------------
__global__ void convert_wout(const float* __restrict__ w, _Float16* __restrict__ o, int n) {
    int i = blockIdx.x * blockDim.x + threadIdx.x;
    if (i < n) o[i] = (_Float16)w[i];
}

// ---------------- kernel 2: Y = X @ W^T (+bias), fp32 ----------------
// X: (nrows, 512)  W: (512 out, 512 in) row-major  Y: (nrows, 512)
__global__ void __launch_bounds__(256) proj16(const float* __restrict__ X,
                                              const float* __restrict__ W,
                                              const float* __restrict__ bias,
                                              float* __restrict__ Y, int nrows) {
    __shared__ float Xs[16 * JDIM];   // transposed: Xs[k*16 + r]
    const int tid = threadIdx.x;
    const int r0  = blockIdx.x * 16;
    for (int i = tid; i < 16 * JDIM; i += 256) {
        int k = i & (JDIM - 1), r = i >> 9;
        int row = r0 + r;
        Xs[k * 16 + r] = (row < nrows) ? X[(size_t)row * JDIM + k] : 0.0f;
    }
    __syncthreads();
    for (int jj = 0; jj < 2; ++jj) {
        int j = tid + jj * 256;
        const float* wr = W + (size_t)j * JDIM;
        float acc[16];
#pragma unroll
        for (int r = 0; r < 16; ++r) acc[r] = 0.0f;
        for (int k = 0; k < JDIM; ++k) {
            float wv = wr[k];
            const float* xc = &Xs[k * 16];
#pragma unroll
            for (int r = 0; r < 16; ++r) acc[r] = fmaf(xc[r], wv, acc[r]);
        }
        float bv = bias ? bias[j] : 0.0f;
        for (int r = 0; r < 16; ++r) {
            int row = r0 + r;
            if (row < nrows) Y[(size_t)row * JDIM + j] = acc[r] + bv;
        }
    }
}

// ---------------- kernel 3: fused tanh(enc+dec) @ W_out^T + b_out ----------------
// grid: (BT, 4 V-slices of 256 cols), block: 256 threads (8 waves).
// Wave w: mPair = w&1 -> M tiles {2*mPair, 2*mPair+1} (32 u rows),
//         nQuad = w>>1 -> N cols [nQuad*64, +64) (4 N-tiles).
// 8 accumulators/wave; B global traffic 2x-shared, A comes from LDS (cheap).
__global__ void __launch_bounds__(256) joint_main(const float* __restrict__ enc_proj, // (BT, 512)
                                                  const float* __restrict__ dec_proj, // (B*U, 512)
                                                  const _Float16* __restrict__ Wh,    // (V, 512) f16
                                                  const float* __restrict__ b_out,    // (V)
                                                  float* __restrict__ out)            // (BT, U, V)
{
    extern __shared__ _Float16 Hs[];          // UDIM * HS_STRIDE halves (~65 KB)
    const int bt    = blockIdx.x;
    const int vbase = blockIdx.y * 256;
    const int b     = bt >> 8;                // T = 256
    const int tid   = threadIdx.x;

    // Build H = tanh(enc_row + dec_rows) once per block, f16 into LDS.
    const float* encRow  = enc_proj + (size_t)bt * JDIM;
    const float* decBase = dec_proj + (size_t)b * UDIM * JDIM;
    for (int i = tid; i < UDIM * JDIM; i += 256) {
        int u = i >> 9, k = i & (JDIM - 1);
        float h = tanhf(encRow[k] + decBase[u * JDIM + k]);
        Hs[u * HS_STRIDE + k] = (_Float16)h;
    }
    __syncthreads();

    const int wave   = tid >> 5;
    const int lane   = tid & 31;
    const int mPair  = wave & 1;      // M tiles 2*mPair, 2*mPair+1
    const int nQuad  = wave >> 1;     // N cols [nQuad*64, +64)
    const int laneN  = lane & 15;
    const int hiHalf = lane >> 4;     // 0 or 1
    const int kShift = hiHalf * 8;    // ISA 16-bit A/B fragment K split

    f32x8 acc[2][4];
#pragma unroll
    for (int mt = 0; mt < 2; ++mt)
#pragma unroll
        for (int nt = 0; nt < 4; ++nt)
#pragma unroll
            for (int j = 0; j < 8; ++j) acc[mt][nt][j] = 0.0f;

    // Hoisted base pointers: all K offsets (k0*2 + {0,32} bytes <= 1056) become
    // immediate offsets on ds_load_b128 / global_load_b128.
    const _Float16* aRow0 = Hs + ((mPair * 2 + 0) * 16 + laneN) * HS_STRIDE + kShift;
    const _Float16* aRow1 = Hs + ((mPair * 2 + 1) * 16 + laneN) * HS_STRIDE + kShift;
    const int colBase = vbase + nQuad * 64 + laneN;
    const _Float16* bPtr0 = Wh + (size_t)(colBase +  0) * JDIM + kShift;
    const _Float16* bPtr1 = Wh + (size_t)(colBase + 16) * JDIM + kShift;
    const _Float16* bPtr2 = Wh + (size_t)(colBase + 32) * JDIM + kShift;
    const _Float16* bPtr3 = Wh + (size_t)(colBase + 48) * JDIM + kShift;
    const _Float16* bPtrs[4] = { bPtr0, bPtr1, bPtr2, bPtr3 };

#pragma unroll
    for (int k0 = 0; k0 < JDIM; k0 += 32) {
        union { half16 v; half8 h[2]; } a0, a1;
        a0.h[0] = *(const half8*)(aRow0 + k0);
        a0.h[1] = *(const half8*)(aRow0 + k0 + 16);
        a1.h[0] = *(const half8*)(aRow1 + k0);
        a1.h[1] = *(const half8*)(aRow1 + k0 + 16);
#pragma unroll
        for (int nt = 0; nt < 4; ++nt) {
            union { half16 v; half8 h[2]; } bf;
            bf.h[0] = *(const half8*)(bPtrs[nt] + k0);
            bf.h[1] = *(const half8*)(bPtrs[nt] + k0 + 16);
            acc[0][nt] = __builtin_amdgcn_wmma_f32_16x16x32_f16(
                false, a0.v, false, bf.v, (short)0, acc[0][nt], false, false);
            acc[1][nt] = __builtin_amdgcn_wmma_f32_16x16x32_f16(
                false, a1.v, false, bf.v, (short)0, acc[1][nt], false, false);
        }
    }

    // Epilogue: add bias, store fp32 per C/D layout (VGPR r -> M = r + 8*hiHalf, N = laneN).
    float* outRow = out + (size_t)bt * UDIM * VDIM;
#pragma unroll
    for (int nt = 0; nt < 4; ++nt) {
        int v = colBase + nt * 16;
        float bv = b_out[v];
#pragma unroll
        for (int mt = 0; mt < 2; ++mt) {
#pragma unroll
            for (int r = 0; r < 8; ++r) {
                int u = (mPair * 2 + mt) * 16 + hiHalf * 8 + r;
                outRow[(size_t)u * VDIM + v] = acc[mt][nt][r] + bv;
            }
        }
    }
}

extern "C" void kernel_launch(void* const* d_in, const int* in_sizes, int n_in,
                              void* d_out, int out_size, void* d_ws, size_t ws_size,
                              hipStream_t stream) {
    const float* encoder_out = (const float*)d_in[0];  // (4,256,1,512)
    const float* decoder_out = (const float*)d_in[1];  // (4,1,64,512)
    const float* W_enc       = (const float*)d_in[2];  // (512,512)
    const float* b_enc       = (const float*)d_in[3];  // (512)
    const float* W_dec       = (const float*)d_in[4];  // (512,512)
    const float* W_out       = (const float*)d_in[5];  // (1024,512)
    const float* b_out       = (const float*)d_in[6];  // (1024)
    float* out = (float*)d_out;                        // (4,256,64,1024)

    char* ws = (char*)d_ws;
    float*    enc_proj = (float*)ws;                                        // 2 MB
    float*    dec_proj = (float*)(ws + (size_t)(2 << 20));                  // 512 KB
    _Float16* Wh       = (_Float16*)(ws + (size_t)(2 << 20) + (512 << 10)); // 1 MB

    convert_wout<<<(VDIM * JDIM + 255) / 256, 256, 0, stream>>>(W_out, Wh, VDIM * JDIM);
    proj16<<<BT / 16, 256, 0, stream>>>(encoder_out, W_enc, b_enc, enc_proj, BT);
    proj16<<<(BDIM * UDIM) / 16, 256, 0, stream>>>(decoder_out, W_dec, nullptr, dec_proj, BDIM * UDIM);

    dim3 grid(BT, VDIM / 256);
    size_t shmem = (size_t)UDIM * HS_STRIDE * sizeof(_Float16);  // 66560 B
    joint_main<<<grid, dim3(256), shmem, stream>>>(enc_proj, dec_proj, Wh, b_out, out);
}
